// AgeUGP_v1_30030411334317
// MI455X (gfx1250) — matrix operand order repaired and checked by reading, not compiled, not run on hardware
//
#include <hip/hip_runtime.h>
#include <hip/hip_bf16.h>
#include <math.h>

// ---------------------------------------------------------------------------
// AgeUGP forward on MI455X (gfx1250, wave32).
//   Stage 1: fmean[s] = mean_f filters[f,s]
//   Stage 2: sample_h[b,g] = sum_{seg[n]==g} snp[b,ids[n]]*fmean[ids[n]]
//            (sorted segments -> run-length accumulate, atomic per run)
//   Stage 3: h1 = sample_h @ W1 (+b1 folded into stage 4)
//            split-K over 25 blocks; per block: TDM tensor_load_to_lds
//            double-buffered tiles (A 32x32 f32, B 32x64 f32) ->
//            s_wait_tensorcnt -> barrier -> LDS reads -> f16 ->
//            v_wmma_f32_16x16x32_f16 -> f32 atomicAdd partials.
//   Stage 4: one wave (lane==batch) does all BN/MLP/loss math with
//            __shfl_xor cross-lane reductions for batch statistics.
// ---------------------------------------------------------------------------

#define N_SNPS   100000
#define N_GENES  20000
#define N_NODES  200000
#define BATCH    32

#define KSPLIT   25                   // split-K blocks
#define KC_GENES (N_GENES / KSPLIT)   // 800 genes per block
#define KSTEPS   (KC_GENES / 32)      // 25 wmma K-steps per block

typedef __attribute__((ext_vector_type(16))) _Float16 v16h;
typedef __attribute__((ext_vector_type(8)))  float    v8f;
typedef __attribute__((ext_vector_type(4)))  unsigned int v4u;
typedef __attribute__((ext_vector_type(8)))  int      v8i;
typedef __attribute__((ext_vector_type(4)))  int      v4i;

// ---------------- Stage 1: filter mean over NF=8 ----------------
__global__ __launch_bounds__(256) void k_fmean(const float* __restrict__ filters,
                                               float* __restrict__ fmean) {
    int s = blockIdx.x * blockDim.x + threadIdx.x;
    if (s >= N_SNPS) return;
    float a = 0.f;
#pragma unroll
    for (int f = 0; f < 8; ++f) a += filters[f * N_SNPS + s];
    fmean[s] = a * 0.125f;
}

__global__ __launch_bounds__(256) void k_zero(float* __restrict__ p, int n) {
    int i = blockIdx.x * blockDim.x + threadIdx.x;
    if (i < n) p[i] = 0.f;
}

// ---------------- Stage 2: gather + segmented sum ----------------
__global__ __launch_bounds__(256) void k_scatter(const float* __restrict__ snp,
                                                 const int*   __restrict__ snp_ids,
                                                 const int*   __restrict__ seg_ids,
                                                 const float* __restrict__ fmean,
                                                 float*       __restrict__ sampleh) {
    const int NPT = 8;
    const int b  = blockIdx.y;
    int n0 = (blockIdx.x * blockDim.x + threadIdx.x) * NPT;
    if (n0 >= N_NODES) return;
    int nend = n0 + NPT; if (nend > N_NODES) nend = N_NODES;

    const float* srow = snp + (size_t)b * N_SNPS;
    float*       orow = sampleh + (size_t)b * N_GENES;

    int   curg = seg_ids[n0];
    float acc  = 0.f;
    for (int n = n0; n < nend; ++n) {
        int g = seg_ids[n];
        int s = snp_ids[n];
        float v = srow[s] * fmean[s];
        if (g != curg) { atomicAdd(orow + curg, acc); acc = 0.f; curg = g; }
        acc += v;
    }
    atomicAdd(orow + curg, acc);
}

// ---------------- TDM helper: 2-D f32 tile load into LDS ----------------
// Packs a Tensor DMA Descriptor (ISA cdna5 8.3/8.4): group0 = flags/lds/global
// address/type, group1 = data_size + tensor dims + tile dims + dim0 stride.
// 2-D tensor -> groups 2/3 zero. All operands are wave-uniform (SGPR-sourced).
__device__ __forceinline__ void tdm_load_2d_f32(unsigned lds_byte,
                                                const float* gaddr,
                                                unsigned tensor_d0,  // row len (elems)
                                                unsigned tensor_d1,  // #rows
                                                unsigned tile_d0,    // tile row len
                                                unsigned tile_d1,    // tile #rows
                                                unsigned long long stride0) { // elems
    unsigned long long ga = (unsigned long long)(uintptr_t)gaddr;
    v4u g0;
    g0[0] = 1u;                                   // count=1, is_restore=0, no gather
    g0[1] = lds_byte;                             // D#.lds_addr (bytes)
    g0[2] = (unsigned)(ga & 0xffffffffu);         // global_addr[31:0]
    g0[3] = (unsigned)((ga >> 32) & 0x01ffffffu)  // global_addr[56:32]
          | (2u << 30);                           // type = 2 ("image")
    unsigned w0 = (2u << 16);                     // data_size=2 -> 4-byte elems
    unsigned w1 = (tensor_d0 & 0xffffu) << 16;    // tensor_dim0[15:0] @ bits63:48
    unsigned w2 = (tensor_d0 >> 16) | ((tensor_d1 & 0xffffu) << 16);
    unsigned w3 = (tensor_d1 >> 16) | (tile_d0 << 16);
    unsigned w4 = (tile_d1 & 0xffffu);            // tile_dim1 (tile_dim2 = 0)
    unsigned w5 = (unsigned)(stride0 & 0xffffffffu);
    unsigned w6 = (unsigned)((stride0 >> 32) & 0xffffu);
    v8i g1 = {(int)w0, (int)w1, (int)w2, (int)w3, (int)w4, (int)w5, (int)w6, 0};
    v4i z4 = {0, 0, 0, 0};
#if defined(__clang_major__) && (__clang_major__ >= 23)
    v8i z8 = {0, 0, 0, 0, 0, 0, 0, 0};
    __builtin_amdgcn_tensor_load_to_lds(g0, g1, z4, z4, z8, 0);
#else
    __builtin_amdgcn_tensor_load_to_lds(g0, g1, z4, z4, 0);
#endif
}

// ---------------- Stage 3: split-K WMMA GEMM with TDM-fed LDS tiles -------
// Grid: KSPLIT blocks x 256 threads (8 waves). Block kb owns genes
// [kb*800, kb*800+800). Wave w -> output tile (mt=(w>>2)*16, nt=(w&3)*16).
// Per K-step: wave0 DMA-loads A tile (32 rows x 32 k, row stride 20000) and
// B tile (32 k x 64 n, contiguous) into the next LDS buffer, waits
// TENSORcnt, block barrier, everyone computes one WMMA from LDS.
__global__ __launch_bounds__(256) void k_gemm(const float* __restrict__ S,   // [32,20000]
                                              const float* __restrict__ W1,  // [20000,64]
                                              float* __restrict__ H1) {      // [32,64] partials
    // LDS: A double buffer (2 x 1024 f32) then B double buffer (2 x 2048 f32).
    // Single shared array => byte offsets below match D#.lds_addr (LDS base 0).
    // NOTE: no pointer arrays into LDS (addrspacecast static-init is unlinkable);
    // buffer selection is pure pointer arithmetic.
    __shared__ float smem[2 * 1024 + 2 * 2048];

    const int wave = threadIdx.x >> 5;
    const int lane = threadIdx.x & 31;
    const int mt   = (wave >> 2) * 16;
    const int nt   = (wave & 3) * 16;
    const int l15  = lane & 15;
    const bool hi  = lane >= 16;
    const int akbase = hi ? 8 : 0;     // A 16-bit 16x32 lane layout (ISA 7.12.2)
    const int bkbase = hi ? 8 : 0;     // B halfs: same K order as A

    const int kc0 = blockIdx.x * KC_GENES;

    // prologue: issue step 0 tiles (TDM ignores EXEC; one issue per wave 0)
    if (wave == 0) {
        tdm_load_2d_f32(/*lds*/ 0u, S + kc0, N_GENES, 32, 32, 32, N_GENES);
        tdm_load_2d_f32(/*lds*/ 8192u, W1 + (size_t)kc0 * 64, 64, N_GENES, 64, 32, 64);
    }

    v8f acc = {};
    for (int i = 0; i < KSTEPS; ++i) {
        const int cur = i & 1;
        if (wave == 0) {
            if (i + 1 < KSTEPS) {
                const unsigned nxt = (unsigned)((i + 1) & 1);
                const int ks = kc0 + (i + 1) * 32;
                tdm_load_2d_f32(nxt * 4096u, S + ks, N_GENES, 32, 32, 32, N_GENES);
                tdm_load_2d_f32(8192u + nxt * 8192u, W1 + (size_t)ks * 64,
                                64, N_GENES, 64, 32, 64);
                // 2 newly-issued may remain in flight; step-i pair must be done
                __builtin_amdgcn_s_wait_tensorcnt(2);
            } else {
                __builtin_amdgcn_s_wait_tensorcnt(0);
            }
        }
        __syncthreads();   // publish TDM-written tile to all 8 waves

        const float* A = smem + cur * 1024;          // LDS byte 0 / 4096
        const float* B = smem + 2048 + cur * 2048;   // LDS byte 8192 / 16384
        v16h a, b;
        // A: row = mt + l15; local k = akbase+{0..7} and akbase+16+{0..7}
        const float* ar = A + (mt + l15) * 32 + akbase;
        float4 a0 = *(const float4*)(ar);
        float4 a1 = *(const float4*)(ar + 4);
        float4 a2 = *(const float4*)(ar + 16);
        float4 a3 = *(const float4*)(ar + 20);
        a[0]=(_Float16)a0.x; a[1]=(_Float16)a0.y; a[2]=(_Float16)a0.z; a[3]=(_Float16)a0.w;
        a[4]=(_Float16)a1.x; a[5]=(_Float16)a1.y; a[6]=(_Float16)a1.z; a[7]=(_Float16)a1.w;
        a[8]=(_Float16)a2.x; a[9]=(_Float16)a2.y; a[10]=(_Float16)a2.z; a[11]=(_Float16)a2.w;
        a[12]=(_Float16)a3.x; a[13]=(_Float16)a3.y; a[14]=(_Float16)a3.z; a[15]=(_Float16)a3.w;
        // B: half t <-> local k = bkbase + t (+8 for t>=8), column n = nt + l15
#pragma unroll
        for (int t = 0; t < 16; ++t)
            b[t] = (_Float16)B[(bkbase + t + (t >= 8 ? 8 : 0)) * 64 + nt + l15];

        acc = __builtin_amdgcn_wmma_f32_16x16x32_f16(
            false, a, false, b, (short)0, acc, false, false);

        __syncthreads();   // all reads done before next TDM overwrites buffer
    }

    // C layout: VGPR r -> (M = mt + r + (hi?8:0), N = nt + l15); split-K partial
#pragma unroll
    for (int r = 0; r < 8; ++r) {
        int m = mt + r + (hi ? 8 : 0);
        int n = nt + l15;
        atomicAdd(&H1[m * 64 + n], acc[r]);
    }
}

// ---------------- Stage 4: head (single wave, lane == batch) ----------------
__device__ __forceinline__ float wsum(float x) {
#pragma unroll
    for (int m = 16; m >= 1; m >>= 1) x += __shfl_xor(x, m, 32);
    return x;
}
__device__ __forceinline__ float bn_relu(float x, float g, float be) {
    float mu = wsum(x) * (1.f / 32.f);
    float d  = x - mu;
    float var = wsum(d * d) * (1.f / 32.f);   // biased var, matches jnp.var
    float y = g * d * rsqrtf(var + 1e-5f) + be;
    return y > 0.f ? y : 0.f;
}

struct HeadParams {
    const float *b1, *g1, *be1;
    const float *W2, *b2, *g2, *be2;
    const float *W3, *b3, *g3, *be3;
    const float *W4, *b4;
    const float *A1, *a1, *A2, *a2;
    const float *P1, *p1, *gp1, *bp1;
    const float *P2, *p2, *gp2, *bp2;
    const float *P3, *p3;
    const float *age;
    const int   *labels;
};

__global__ __launch_bounds__(32) void k_head(const float* __restrict__ H1,
                                             HeadParams P,
                                             float* __restrict__ out) {
    const int b = threadIdx.x;  // lane == batch sample

    // Layer1 bias + BN + ReLU (64)
    float h[64];
#pragma unroll 4
    for (int c = 0; c < 64; ++c)
        h[c] = bn_relu(H1[b * 64 + c] + P.b1[c], P.g1[c], P.be1[c]);

    // Linear 64->16 + BN + ReLU -> features
    float feat[16];
    for (int j = 0; j < 16; ++j) {
        float s = P.b2[j];
        for (int c = 0; c < 64; ++c) s += h[c] * P.W2[c * 16 + j];
        feat[j] = s;
    }
    for (int j = 0; j < 16; ++j) feat[j] = bn_relu(feat[j], P.g2[j], P.be2[j]);

    // predictor tail: 16->4 BN ReLU, 4->1 -> original logit
    float h3[4];
    for (int j = 0; j < 4; ++j) {
        float s = P.b3[j];
        for (int c = 0; c < 16; ++c) s += feat[c] * P.W3[c * 4 + j];
        h3[j] = bn_relu(s, P.g3[j], P.be3[j]);
    }
    float logit = P.b4[0];
    for (int c = 0; c < 4; ++c) logit += h3[c] * P.W4[c];

    // adversarial age head (GRL = identity forward)
    float a8[8];
    for (int j = 0; j < 8; ++j) {
        float s = P.p1[j];
        for (int c = 0; c < 16; ++c) s += feat[c] * P.P1[c * 8 + j];
        a8[j] = bn_relu(s, P.gp1[j], P.bp1[j]);
    }
    float a4[4];
    for (int j = 0; j < 4; ++j) {
        float s = P.p2[j];
        for (int c = 0; c < 8; ++c) s += a8[c] * P.P2[c * 4 + j];
        a4[j] = bn_relu(s, P.gp2[j], P.bp2[j]);
    }
    float ap = P.p3[0];
    for (int c = 0; c < 4; ++c) ap += a4[c] * P.P3[c];
    float age_pred = 1.f / (1.f + expf(-ap));

    float age_norm = (P.age[b] - 40.f) * (1.f / 30.f);
    float age_loss = wsum(fabsf(age_pred - age_norm)) * (1.f / 32.f);

    // transition from age: relu(age_norm @ A1 + a1) @ A2 + a2 -> softmax
    float t0 = P.a2[0], t1 = P.a2[1];
    for (int k = 0; k < 8; ++k) {
        float r = age_norm * P.A1[k] + P.a1[k];
        r = r > 0.f ? r : 0.f;
        t0 += r * P.A2[k * 2 + 0];
        t1 += r * P.A2[k * 2 + 1];
    }
    float mx = t0 > t1 ? t0 : t1;
    float e0 = expf(t0 - mx), e1 = expf(t1 - mx);
    float pos1 = e1 / (e0 + e1);                       // pos_probs[:,1]

    float probs = 1.f / (1.f + expf(-logit));
    probs = fminf(fmaxf(probs, 1e-7f), 1.f - 1e-7f);
    float updated = probs * pos1;   // einsum col 1 == probs * pos_probs[:,1]

    float bce  = -(probs * logf(updated) + (1.f - probs) * logf(1.f - updated));
    float mask = (P.labels[b] == 1) ? 1.f : 0.f;
    float cons = wsum(mask * bce) / fmaxf(wsum(mask), 1.f);

    float finalp = (mask > 0.f) ? probs : updated;
    float final_logit = logf(finalp / (1.f - finalp + 1e-7f));

    out[b]         = final_logit;    // final_logits [32]
    out[BATCH + b] = logit;          // original_logits [32]
    if (b == 0) out[2 * BATCH] = cons + 0.5f * age_loss;   // final_loss
}

// ---------------------------------------------------------------------------
extern "C" void kernel_launch(void* const* d_in, const int* in_sizes, int n_in,
                              void* d_out, int out_size, void* d_ws, size_t ws_size,
                              hipStream_t stream) {
    // setup_inputs() order with params pytree flattened in JAX sorted-key order:
    // 0:snp 1:age 2:A1 3:A2 4:P1 5:P2 6:P3 7:W1 8:W2 9:W3 10:W4 11:a1 12:a2
    // 13:b1 14:b2 15:b3 16:b4 17:be1 18:be2 19:be3 20:bp1 21:bp2 22:filters
    // 23:g1 24:g2 25:g3 26:gp1 27:gp2 28:p1 29:p2 30:p3 31:snp_ids
    // 32:segment_ids 33:labels
    const float* snp     = (const float*)d_in[0];
    const float* age     = (const float*)d_in[1];
    const float* A1      = (const float*)d_in[2];
    const float* A2      = (const float*)d_in[3];
    const float* P1      = (const float*)d_in[4];
    const float* P2      = (const float*)d_in[5];
    const float* P3      = (const float*)d_in[6];
    const float* W1      = (const float*)d_in[7];
    const float* W2      = (const float*)d_in[8];
    const float* W3      = (const float*)d_in[9];
    const float* W4      = (const float*)d_in[10];
    const float* a1      = (const float*)d_in[11];
    const float* a2      = (const float*)d_in[12];
    const float* b1      = (const float*)d_in[13];
    const float* b2      = (const float*)d_in[14];
    const float* b3      = (const float*)d_in[15];
    const float* b4      = (const float*)d_in[16];
    const float* be1     = (const float*)d_in[17];
    const float* be2     = (const float*)d_in[18];
    const float* be3     = (const float*)d_in[19];
    const float* bp1     = (const float*)d_in[20];
    const float* bp2     = (const float*)d_in[21];
    const float* filters = (const float*)d_in[22];
    const float* g1      = (const float*)d_in[23];
    const float* g2      = (const float*)d_in[24];
    const float* g3      = (const float*)d_in[25];
    const float* gp1     = (const float*)d_in[26];
    const float* gp2     = (const float*)d_in[27];
    const float* p1      = (const float*)d_in[28];
    const float* p2      = (const float*)d_in[29];
    const float* p3      = (const float*)d_in[30];
    const int* snp_ids   = (const int*)d_in[31];
    const int* seg_ids   = (const int*)d_in[32];
    const int* labels    = (const int*)d_in[33];
    float* out = (float*)d_out;

    // workspace layout
    char* ws = (char*)d_ws;
    float* fmean   = (float*)(ws);                       // 100000 f32
    float* sampleh = (float*)(ws + 400000);              // 32*20000 f32
    float* h1      = (float*)(ws + 400000 + 2560000);    // 32*64 f32 (contiguous after)

    // Stage 1
    k_fmean<<<(N_SNPS + 255) / 256, 256, 0, stream>>>(filters, fmean);
    // zero sample_h AND h1 accumulators every call (both atomically built)
    k_zero<<<(BATCH * N_GENES + BATCH * 64 + 255) / 256, 256, 0, stream>>>(
        sampleh, BATCH * N_GENES + BATCH * 64);
    // Stage 2: gather + segmented sum
    {
        const int NPT = 8;
        int threads_per_batch = (N_NODES + NPT - 1) / NPT;  // 25000
        dim3 grid((threads_per_batch + 255) / 256, BATCH, 1);
        k_scatter<<<grid, 256, 0, stream>>>(snp, snp_ids, seg_ids, fmean, sampleh);
    }
    // Stage 3: split-K WMMA GEMM, TDM double-buffered tiles
    k_gemm<<<KSPLIT, 256, 0, stream>>>(sampleh, W1, h1);
    // Stage 4: head on one wave
    HeadParams hp = { b1, g1, be1, W2, b2, g2, be2, W3, b3, g3, be3, W4, b4,
                      A1, a1, A2, a2, P1, p1, gp1, bp1, P2, p2, gp2, bp2,
                      P3, p3, age, labels };
    k_head<<<1, 32, 0, stream>>>(h1, hp, out);
}
